// density_wfa_finetune_20804821582460
// MI455X (gfx1250) — compile-verified, matrix operand order checked by reading, not run on hardware
//
#include <hip/hip_runtime.h>
#include <hip/hip_bf16.h>

#define N_TOT   8192
#define L_SEQ   128
#define KMIX    10

#define WAVES       4
#define ROWS_WAVE   32
#define ROWS_BLOCK  128
#define NBLOCKS     (N_TOT / ROWS_BLOCK)

// fragment indices inside d_ws (each fragment = 32 lanes * 32 bytes = 1 KB)
#define FR_A      0      // 512 frags: A reshaped (4096 x 64)
#define FR_ENC2   512    // 8 frags:  enc2_w (64 x 64)
#define FR_NADE   520    // 16 frags: nade_w (64 x 128)
#define FR_MU     536    // 4 frags:  mu_w (128 x 10, j padded to 16)
#define FR_SIG    540    // 4
#define FR_AL     544    // 4
#define FR_TOTAL  548

// dynamic LDS layout (bytes)
#define AB_BYTES   65536                       // one A chunk (8 i-values)
#define TMPS_OFF   (2*AB_BYTES)                // per-wave state as duplicated bf16 pairs, 32x64 u32
#define STAGE_OFF  (TMPS_OFF + WAVES*8192)     // per-wave bf16 transpose stage, 32x128
#define CONST_OFF  (STAGE_OFF + WAVES*8192)
#define SMEM_BYTES (CONST_OFF + 2048)

typedef __attribute__((ext_vector_type(16))) __bf16  v16bf;
typedef __attribute__((ext_vector_type(8)))  float   v8f;
typedef __attribute__((ext_vector_type(4)))  unsigned u32x4;
typedef __attribute__((ext_vector_type(8)))  int     i32x8;
typedef __attribute__((ext_vector_type(4)))  int     i32x4;

union Frag { v16bf v; u32x4 q[2]; unsigned u[8]; unsigned short h[16]; };

__device__ __forceinline__ unsigned short f2bf(float f) {     // cold path (pack kernel)
  unsigned u = __float_as_uint(f);
  return (unsigned short)((u + 0x7fffu + ((u >> 16) & 1u)) >> 16);  // RNE
}

// hardware packed convert: dst = {bf16(b), bf16(a)}  (1 VALU)
__device__ __forceinline__ unsigned cvtpk_bf16(float a, float b) {
  unsigned d;
  asm("v_cvt_pk_bf16_f32 %0, %1, %2" : "=v"(d) : "v"(a), "v"(b));
  return d;
}

// branch-free tanh: 1 - 2/(1+e^{2x});  exact at +/-inf, ~5 VALU, no EXEC games
__device__ __forceinline__ float fast_tanh(float x) {
  float u = __expf(x + x);
  return fmaf(-2.f, __builtin_amdgcn_rcpf(1.f + u), 1.f);
}

__device__ __forceinline__ v8f wmma_bf16(v16bf a, v16bf b, v8f c) {
  return __builtin_amdgcn_wmma_f32_16x16x32_bf16(false, a, false, b, (short)0, c, false, false);
}

// B fragment from prepacked global (ws)
__device__ __forceinline__ v16bf loadBG(const u32x4* __restrict__ ws, int fragIdx, int lane) {
  Frag f;
  f.q[0] = ws[fragIdx * 64 + lane * 2 + 0];
  f.q[1] = ws[fragIdx * 64 + lane * 2 + 1];
  return f.v;
}
// B fragment from LDS chunk buffer
__device__ __forceinline__ v16bf loadBL(const char* Ab, int fragIdx, int lane) {
  const u32x4* p = (const u32x4*)(Ab + fragIdx * 1024 + lane * 32);
  Frag f; f.q[0] = p[0]; f.q[1] = p[1]; return f.v;
}
// A fragment gathered from a bf16 row-major stage buffer (16-bit A layout: lane<16 -> K {0..7,16..23})
__device__ __forceinline__ v16bf loadA(const unsigned short* stg, int ldm, int mt, int kc, int lane) {
  int l15 = lane & 15;
  int kb  = ((lane >> 4) ? 8 : 0) + kc * 32;
  int row = mt * 16 + l15;
  const u32x4* p0 = (const u32x4*)(stg + row * ldm + kb);
  const u32x4* p1 = (const u32x4*)(stg + row * ldm + kb + 16);
  Frag f; f.q[0] = *p0; f.q[1] = *p1; return f.v;
}

__device__ __forceinline__ float redmax16(float v) {
  #pragma unroll
  for (int s = 1; s < 16; s <<= 1) v = fmaxf(v, __shfl_xor(v, s, 16));
  return v;
}
__device__ __forceinline__ float redsum16(float v) {
  #pragma unroll
  for (int s = 1; s < 16; s <<= 1) v += __shfl_xor(v, s, 16);
  return v;
}

// TDM: DMA a contiguous 64KB block (8192 x 8B elements, 1-D tile) from global into LDS.
__device__ __forceinline__ void tdm_issue(const void* gsrc, unsigned lds_off) {
  unsigned long long ga = (unsigned long long)gsrc;
  u32x4 g0;
  g0[0] = 1u;                                    // count=1, user descriptor
  g0[1] = lds_off;                               // lds_addr
  g0[2] = (unsigned)ga;                          // global_addr[31:0]
  g0[3] = (unsigned)(ga >> 32) | (2u << 30);     // global_addr[56:32] | type=2
  i32x8 g1;
  g1[0] = (int)(3u << 16);                       // data_size = 8B
  g1[1] = (int)(8192u << 16);                    // tensor_dim0 = 8192 (low 16 at bits 63:48)
  g1[2] = (int)(1u << 16);                       // tensor_dim1 = 1
  g1[3] = (int)(8192u << 16);                    // tile_dim0 = 8192
  g1[4] = 0;                                     // tile_dim1/2 unused
  g1[5] = 8192;                                  // tensor_dim0_stride
  g1[6] = 0; g1[7] = 0;
  i32x4 z4; z4[0] = z4[1] = z4[2] = z4[3] = 0;
  i32x8 z8;
  #pragma unroll
  for (int i = 0; i < 8; ++i) z8[i] = 0;
  __builtin_amdgcn_tensor_load_to_lds(g0, g1, z4, z4, z8, 0);
}

// ---------------------------------------------------------------------------
// Kernel 0: prepack all weight matrices into bf16 WMMA-B fragment layout.
// ---------------------------------------------------------------------------
__global__ void __launch_bounds__(128, 1) pack_kernel(
    const float* __restrict__ A, const float* __restrict__ enc2_w,
    const float* __restrict__ nade_w, const float* __restrict__ mu_w,
    const float* __restrict__ sig_w, const float* __restrict__ alpha_w,
    u32x4* __restrict__ ws)
{
  int tid  = blockIdx.x * 128 + threadIdx.x;
  int frag = tid >> 5, lane = tid & 31;
  if (frag >= FR_TOTAL) return;

  const float* src; int Kdim, Jdim, kc, jt;
  if (frag < FR_ENC2)      { src = A;       Kdim = 4096; Jdim = 64;  kc = frag >> 2;            jt = frag & 3; }
  else if (frag < FR_NADE) { src = enc2_w;  Kdim = 64;   Jdim = 64;  int f = frag - FR_ENC2;    kc = f >> 2; jt = f & 3; }
  else if (frag < FR_MU)   { src = nade_w;  Kdim = 64;   Jdim = 128; int f = frag - FR_NADE;    kc = f >> 3; jt = f & 7; }
  else if (frag < FR_SIG)  { src = mu_w;    Kdim = 128;  Jdim = 10;  kc = frag - FR_MU;         jt = 0; }
  else if (frag < FR_AL)   { src = sig_w;   Kdim = 128;  Jdim = 10;  kc = frag - FR_SIG;        jt = 0; }
  else                     { src = alpha_w; Kdim = 128;  Jdim = 10;  kc = frag - FR_AL;         jt = 0; }

  int l15 = lane & 15;
  int kb  = ((lane >> 4) ? 8 : 0) + kc * 32;
  int j   = jt * 16 + l15;
  Frag f;
  #pragma unroll
  for (int v = 0; v < 16; ++v) {
    int k = kb + (v < 8 ? v : v + 8);
    float val = (j < Jdim && k < Kdim) ? src[k * Jdim + j] : 0.f;
    f.h[v] = f2bf(val);
  }
  ws[frag * 64 + lane * 2 + 0] = f.q[0];
  ws[frag * 64 + lane * 2 + 1] = f.q[1];
}

// ---------------------------------------------------------------------------
// Kernel 1: full sequential scan. One wave owns 32 rows for all 128 steps.
// ---------------------------------------------------------------------------
__global__ void __launch_bounds__(WAVES * 32, 1) scan_kernel(
    const float* __restrict__ X,
    const float* __restrict__ w1, const float* __restrict__ b1,
    const float* __restrict__ b2, const float* __restrict__ init_w,
    const float* __restrict__ nade_b, const float* __restrict__ mu_b,
    const float* __restrict__ sig_b, const float* __restrict__ alpha_b,
    const u32x4* __restrict__ ws, float* __restrict__ out)
{
  extern __shared__ char smem[];
  const int tid  = threadIdx.x;
  const int lane = tid & 31, wave = tid >> 5;
  const int l15  = lane & 15, hi = lane >> 4;
  const int rowBase = blockIdx.x * ROWS_BLOCK + wave * ROWS_WAVE;

  unsigned*       tmpS  = (unsigned*)(smem + TMPS_OFF + wave * 8192);       // dup'd bf16 pairs
  unsigned short* stage = (unsigned short*)(smem + STAGE_OFF + wave * 8192);
  float*          cst   = (float*)(smem + CONST_OFF);

  if (tid < 64) { cst[tid] = w1[tid]; cst[64 + tid] = b1[tid]; cst[128 + tid] = b2[tid]; }
  if (tid < 128) cst[192 + tid] = nade_b[tid];
  if (tid < 16) {
    cst[320 + tid] = (tid < KMIX) ? mu_b[tid]    : 0.f;
    cst[336 + tid] = (tid < KMIX) ? sig_b[tid]   : 0.f;
    cst[352 + tid] = (tid < KMIX) ? alpha_b[tid] : 0.f;
  }
  __syncthreads();

  // state (WMMA C layout): acc[mt][jt] covers rows mt*16..+15, cols jt*16..+15
  v8f acc[2][4];
  #pragma unroll
  for (int jt = 0; jt < 4; ++jt) {
    float iw = init_w[jt * 16 + l15];
    #pragma unroll
    for (int mt = 0; mt < 2; ++mt)
      #pragma unroll
      for (int r = 0; r < 8; ++r) acc[mt][jt][r] = iw;
  }
  float logp[2][8];
  #pragma unroll
  for (int mt = 0; mt < 2; ++mt)
    #pragma unroll
    for (int r = 0; r < 8; ++r) logp[mt][r] = 0.f;

  const unsigned ldsBase = (unsigned)(unsigned long long)smem;
  if (wave == 0) tdm_issue((const char*)ws, ldsBase);   // preload A chunk 0 -> buf 0

  #pragma unroll 1
  for (int t = 0; t < L_SEQ; ++t) {
    if (t > 0) {
      // ---- encoder for x_{t-1}: e = tanh(relu(x*w1+b1) @ W2 + b2) ----
      #pragma unroll
      for (int mt = 0; mt < 2; ++mt) {
        float xp = X[(rowBase + mt * 16 + l15) * L_SEQ + (t - 1)];
        Frag H[2];
        #pragma unroll
        for (int kc = 0; kc < 2; ++kc)
          #pragma unroll
          for (int v = 0; v < 16; v += 2) {
            int k0 = (hi ? 8 : 0) + kc * 32 + (v     < 8 ? v     : v + 8);
            int k1 = (hi ? 8 : 0) + kc * 32 + (v + 1 < 8 ? v + 1 : v + 9);
            float h0 = fmaxf(fmaf(xp, cst[k0], cst[64 + k0]), 0.f);
            float h1 = fmaxf(fmaf(xp, cst[k1], cst[64 + k1]), 0.f);
            H[kc].u[v >> 1] = cvtpk_bf16(h0, h1);
          }
        #pragma unroll
        for (int jt = 0; jt < 4; ++jt) {
          v8f e; float bb = cst[128 + jt * 16 + l15];
          #pragma unroll
          for (int r = 0; r < 8; ++r) e[r] = bb;
          e = wmma_bf16(H[0].v, loadBG(ws, FR_ENC2 + jt,     lane), e);
          e = wmma_bf16(H[1].v, loadBG(ws, FR_ENC2 + 4 + jt, lane), e);
          #pragma unroll
          for (int r = 0; r < 8; ++r) {
            int row = mt * 16 + (hi ? 8 + r : r);
            float ft = fast_tanh(e[r]);
            stage[row * 64 + jt * 16 + l15] = (unsigned short)cvtpk_bf16(ft, ft);
          }
        }
      }
      Frag ea[2][2];
      #pragma unroll
      for (int mt = 0; mt < 2; ++mt)
        #pragma unroll
        for (int kc = 0; kc < 2; ++kc) { Frag f; f.v = loadA(stage, 64, mt, kc, lane); ea[mt][kc] = f; }

      // ---- bilinear: tmp_new = (tmp (x) e) @ A_r, streamed over 8 chunks of i ----
      #pragma unroll
      for (int mt = 0; mt < 2; ++mt)
        #pragma unroll
        for (int jt = 0; jt < 4; ++jt)
          #pragma unroll
          for (int r = 0; r < 8; ++r) acc[mt][jt][r] = 0.f;

      #pragma unroll 1
      for (int c = 0; c < 8; ++c) {
        if (wave == 0) __builtin_amdgcn_s_wait_tensorcnt(0);   // chunk c resident
        __syncthreads();
        if (wave == 0 && !(t == L_SEQ - 1 && c == 7))
          tdm_issue((const char*)ws + (((c + 1) & 7) * AB_BYTES),
                    ldsBase + (((c + 1) & 1) * AB_BYTES));
        const char* Ab = smem + ((c & 1) * AB_BYTES);
        #pragma unroll
        for (int ii = 0; ii < 8; ++ii) {
          int i = c * 8 + ii;
          // A-matrix B-fragments are shared by both M-tiles: load once per i
          Frag bf[8];
          #pragma unroll
          for (int f8 = 0; f8 < 8; ++f8) { Frag f; f.v = loadBL(Ab, ii * 8 + f8, lane); bf[f8] = f; }
          #pragma unroll
          for (int mt = 0; mt < 2; ++mt) {
            unsigned tp = tmpS[(mt * 16 + l15) * 64 + i];   // pre-dup'd bf16 pair tmp[row,i]
            Frag a0, a1;
            #pragma unroll
            for (int q = 0; q < 8; ++q) {
              asm("v_pk_mul_bf16 %0, %1, %2" : "=v"(a0.u[q]) : "v"(ea[mt][0].u[q]), "v"(tp));
              asm("v_pk_mul_bf16 %0, %1, %2" : "=v"(a1.u[q]) : "v"(ea[mt][1].u[q]), "v"(tp));
            }
            #pragma unroll
            for (int jt = 0; jt < 4; ++jt) {
              acc[mt][jt] = wmma_bf16(a0.v, bf[jt].v,     acc[mt][jt]);
              acc[mt][jt] = wmma_bf16(a1.v, bf[4 + jt].v, acc[mt][jt]);
            }
          }
        }
        __syncthreads();
      }
    }

    // ---- persist state (as dup'd bf16 pairs) + stage tanh(tmp) as bf16 ----
    #pragma unroll
    for (int mt = 0; mt < 2; ++mt)
      #pragma unroll
      for (int jt = 0; jt < 4; ++jt)
        #pragma unroll
        for (int r = 0; r < 8; ++r) {
          int row = mt * 16 + (hi ? 8 + r : r);
          float v = acc[mt][jt][r];
          tmpS[row * 64 + jt * 16 + l15] = cvtpk_bf16(v, v);
          float ft = fast_tanh(v);
          stage[row * 64 + jt * 16 + l15] = (unsigned short)cvtpk_bf16(ft, ft);
        }
    Frag th[2][2];
    #pragma unroll
    for (int mt = 0; mt < 2; ++mt)
      #pragma unroll
      for (int kc = 0; kc < 2; ++kc) { Frag f; f.v = loadA(stage, 64, mt, kc, lane); th[mt][kc] = f; }

    // ---- NADE hidden: h2 = relu(th @ nade_w + nade_b), staged for head GEMMs ----
    #pragma unroll
    for (int mt = 0; mt < 2; ++mt)
      #pragma unroll
      for (int jt = 0; jt < 8; ++jt) {
        v8f h; float bb = cst[192 + jt * 16 + l15];
        #pragma unroll
        for (int r = 0; r < 8; ++r) h[r] = bb;
        h = wmma_bf16(th[mt][0].v, loadBG(ws, FR_NADE + jt,     lane), h);
        h = wmma_bf16(th[mt][1].v, loadBG(ws, FR_NADE + 8 + jt, lane), h);
        #pragma unroll
        for (int r = 0; r < 8; ++r) {
          int row = mt * 16 + (hi ? 8 + r : r);
          float hr = fmaxf(h[r], 0.f);
          stage[row * 128 + jt * 16 + l15] = (unsigned short)cvtpk_bf16(hr, hr);
        }
      }

    // ---- mixture heads + logsumexp (component k = l15, rows split by lane half) ----
    const bool valid = (l15 < KMIX);
    #pragma unroll
    for (int mt = 0; mt < 2; ++mt) {
      Frag ha[4];
      #pragma unroll
      for (int kc = 0; kc < 4; ++kc) { Frag f; f.v = loadA(stage, 128, mt, kc, lane); ha[kc] = f; }
      v8f mu, ls, al;
      { float m = cst[320 + l15], s = cst[336 + l15], a = cst[352 + l15];
        #pragma unroll
        for (int r = 0; r < 8; ++r) { mu[r] = m; ls[r] = s; al[r] = a; } }
      #pragma unroll
      for (int kc = 0; kc < 4; ++kc) {
        mu = wmma_bf16(ha[kc].v, loadBG(ws, FR_MU  + kc, lane), mu);
        ls = wmma_bf16(ha[kc].v, loadBG(ws, FR_SIG + kc, lane), ls);
        al = wmma_bf16(ha[kc].v, loadBG(ws, FR_AL  + kc, lane), al);
      }
      #pragma unroll
      for (int r = 0; r < 8; ++r) {
        int row = mt * 16 + (hi ? 8 + r : r);
        float x  = X[(rowBase + row) * L_SEQ + t];
        float lg = al[r];
        float m1 = redmax16(valid ? lg : -1e30f);
        float se = redsum16(valid ? __expf(lg - m1) : 0.f);
        float la = lg - m1 - __logf(se);                 // log_softmax
        float lsv = ls[r];
        float z  = (x - mu[r]) * __expf(-lsv);
        float q  = valid ? (la - 0.5f * z * z - lsv - 0.918938533204672742f) : -1e30f;
        float m2 = redmax16(q);
        float s2 = redsum16(valid ? __expf(q - m2) : 0.f);
        logp[mt][r] += m2 + __logf(s2);
      }
    }
  }

  if (l15 == 0) {
    #pragma unroll
    for (int mt = 0; mt < 2; ++mt)
      #pragma unroll
      for (int r = 0; r < 8; ++r)
        out[rowBase + mt * 16 + hi * 8 + r] = __expf(logp[mt][r]);
  }
}

extern "C" void kernel_launch(void* const* d_in, const int* in_sizes, int n_in,
                              void* d_out, int out_size, void* d_ws, size_t ws_size,
                              hipStream_t stream) {
  (void)in_sizes; (void)n_in; (void)out_size; (void)ws_size;
  const float* X       = (const float*)d_in[0];
  const float* enc1_w  = (const float*)d_in[1];
  const float* enc1_b  = (const float*)d_in[2];
  const float* enc2_w  = (const float*)d_in[3];
  const float* enc2_b  = (const float*)d_in[4];
  const float* A       = (const float*)d_in[5];
  const float* init_w  = (const float*)d_in[6];
  const float* nade_w  = (const float*)d_in[7];
  const float* nade_b  = (const float*)d_in[8];
  const float* mu_w    = (const float*)d_in[9];
  const float* mu_b    = (const float*)d_in[10];
  const float* sig_w   = (const float*)d_in[11];
  const float* sig_b   = (const float*)d_in[12];
  const float* alpha_w = (const float*)d_in[13];
  const float* alpha_b = (const float*)d_in[14];

  u32x4* ws = (u32x4*)d_ws;
  int packThreads = FR_TOTAL * 32;
  pack_kernel<<<(packThreads + 127) / 128, 128, 0, stream>>>(A, enc2_w, nade_w, mu_w, sig_w, alpha_w, ws);
  scan_kernel<<<NBLOCKS, WAVES * 32, SMEM_BYTES, stream>>>(
      X, enc1_w, enc1_b, enc2_b, init_w, nade_b, mu_b, sig_b, alpha_b,
      (const u32x4*)d_ws, (float*)d_out);
}
